// Super_Mamba_61400852464211
// MI455X (gfx1250) — matrix operand order repaired and checked by reading, not compiled, not run on hardware
//
#include <hip/hip_runtime.h>
#include <math.h>

typedef __attribute__((ext_vector_type(16))) _Float16 v16h;
typedef __attribute__((ext_vector_type(8)))  float    v8f;

#define SCAN_CS 64   // scan chunk size (scan-time steps per chunk)

// =====================================================================
// WMMA GEMM:  C[M,N] = A[M,K] @ W[N,K]^T  (+bias[N]) (+resid[M,N])
// f32 in/out, f16 WMMA with f32 accumulate.  M % 16 == 0 always here.
// Block = 256 threads = 8 waves; wave w handles M-tile, clamped to M-16
// (duplicate tiles write identical values -> benign).
// K main loop branch-free (float4 -> global_load_b128); K%32 epilogue uses
// clamped-address loads + select (no EXEC divergence around WMMA).
// Fragment layouts per CDNA5 ISA 7.12.2 (wave32).
// =====================================================================
__global__ void k_gemm_wmma(const float* __restrict__ A,
                            const float* __restrict__ W,
                            const float* __restrict__ bias,
                            const float* __restrict__ resid,
                            float* __restrict__ C,
                            int M, int N, int K)
{
    const int lane = threadIdx.x & 31;
    const int wave = threadIdx.x >> 5;

    int tm = blockIdx.x * 128 + wave * 16;
    if (tm + 16 > M) tm = M - 16;            // M%16==0, M>=32 -> safe clamp
    const int tn = blockIdx.y * 16;

    const int arow = tm + (lane & 15);       // A row for this lane
    const int akb  = (lane < 16) ? 0 : 8;    // A K sub-base in 32-tile
    const int bcol = tn + (lane & 15);       // logical N column
    const int brow = (bcol < N) ? bcol : (N - 1);
    const float bm = (bcol < N) ? 1.0f : 0.0f;
    const int bkb  = (lane & 16) ? 8 : 0;    // B K sub-base in 32-tile

    const float* __restrict__ ap = A + (size_t)arow * K;
    const float* __restrict__ wp = W + (size_t)brow * K;

    v8f acc = {};
    const int Kfull = K & ~31;
    int k0 = 0;
    for (; k0 < Kfull; k0 += 32) {
        __builtin_prefetch(ap + k0 + 32, 0, 1);   // global_prefetch_b8
        const float4 a0 = *(const float4*)(ap + k0 + akb);
        const float4 a1 = *(const float4*)(ap + k0 + akb + 4);
        const float4 a2 = *(const float4*)(ap + k0 + akb + 16);
        const float4 a3 = *(const float4*)(ap + k0 + akb + 20);
        const float4 w0 = *(const float4*)(wp + k0 + bkb);
        const float4 w1 = *(const float4*)(wp + k0 + bkb + 4);
        const float4 w2 = *(const float4*)(wp + k0 + bkb + 16);
        const float4 w3 = *(const float4*)(wp + k0 + bkb + 20);

        v16h af, bf;
        af[0]  = (_Float16)a0.x;  af[1]  = (_Float16)a0.y;
        af[2]  = (_Float16)a0.z;  af[3]  = (_Float16)a0.w;
        af[4]  = (_Float16)a1.x;  af[5]  = (_Float16)a1.y;
        af[6]  = (_Float16)a1.z;  af[7]  = (_Float16)a1.w;
        af[8]  = (_Float16)a2.x;  af[9]  = (_Float16)a2.y;
        af[10] = (_Float16)a2.z;  af[11] = (_Float16)a2.w;
        af[12] = (_Float16)a3.x;  af[13] = (_Float16)a3.y;
        af[14] = (_Float16)a3.z;  af[15] = (_Float16)a3.w;

        bf[0]  = (_Float16)(w0.x * bm);  bf[1]  = (_Float16)(w0.y * bm);
        bf[2]  = (_Float16)(w0.z * bm);  bf[3]  = (_Float16)(w0.w * bm);
        bf[4]  = (_Float16)(w1.x * bm);  bf[5]  = (_Float16)(w1.y * bm);
        bf[6]  = (_Float16)(w1.z * bm);  bf[7]  = (_Float16)(w1.w * bm);
        bf[8]  = (_Float16)(w2.x * bm);  bf[9]  = (_Float16)(w2.y * bm);
        bf[10] = (_Float16)(w2.z * bm);  bf[11] = (_Float16)(w2.w * bm);
        bf[12] = (_Float16)(w3.x * bm);  bf[13] = (_Float16)(w3.y * bm);
        bf[14] = (_Float16)(w3.z * bm);  bf[15] = (_Float16)(w3.w * bm);

        acc = __builtin_amdgcn_wmma_f32_16x16x32_f16(
                  false, af, false, bf, (short)0, acc, false, false);
    }
    if (k0 < K) {                            // epilogue tile (K % 32)
        v16h af, bf;
#pragma unroll
        for (int j = 0; j < 8; ++j) {
            int ka = k0 + akb + j;
            int kb = k0 + akb + 16 + j;
            float va = ap[(ka < K) ? ka : (K - 1)] * ((ka < K) ? 1.0f : 0.0f);
            float vb = ap[(kb < K) ? kb : (K - 1)] * ((kb < K) ? 1.0f : 0.0f);
            af[j]     = (_Float16)va;
            af[j + 8] = (_Float16)vb;
        }
#pragma unroll
        for (int j = 0; j < 16; ++j) {
            int kw = k0 + bkb + ((j < 8) ? j : (j + 8));
            float vw = wp[(kw < K) ? kw : (K - 1)] * ((kw < K) ? bm : 0.0f);
            bf[j] = (_Float16)vw;
        }
        acc = __builtin_amdgcn_wmma_f32_16x16x32_f16(
                  false, af, false, bf, (short)0, acc, false, false);
    }

    const int rbase = tm + ((lane & 16) ? 8 : 0);
    const int col   = tn + (lane & 15);
    if (col < N) {
        float badd = bias ? bias[col] : 0.0f;
#pragma unroll
        for (int r = 0; r < 8; ++r) {
            int row = rbase + r;
            float v = acc[r] + badd;
            if (resid) v += resid[(size_t)row * N + col];
            C[(size_t)row * N + col] = v;
        }
    }
}

// ============================ stem conv ==============================
__global__ void k_stem(const float* __restrict__ x, const float* __restrict__ w,
                       const float* __restrict__ b, float* __restrict__ out,
                       int Bn, int H, int Wd)
{
    int idx = blockIdx.x * blockDim.x + threadIdx.x;
    int tot = Bn * H * Wd * 3;
    if (idx >= tot) return;
    int o  = idx % 3;
    int ww = (idx / 3) % Wd;
    int hh = (idx / (3 * Wd)) % H;
    int bb = idx / (3 * Wd * H);
    float acc = b[o];
    for (int i = 0; i < 3; ++i)
        for (int kh = 0; kh < 3; ++kh)
            for (int kw = 0; kw < 3; ++kw) {
                int h2 = hh + kh - 1, w2 = ww + kw - 1;
                if (h2 >= 0 && h2 < H && w2 >= 0 && w2 < Wd)
                    acc += x[(((size_t)bb * 3 + i) * H + h2) * Wd + w2] *
                           w[((o * 3 + i) * 3 + kh) * 3 + kw];
            }
    out[(((size_t)bb * H + hh) * Wd + ww) * 3 + o] = fmaxf(acc, 0.0f);
}

// ========================= patch-merge concat ========================
__global__ void k_pmerge(const float* __restrict__ x, float* __restrict__ y4,
                         int Bn, int Ho, int Wo, int cin)
{
    int idx = blockIdx.x * blockDim.x + threadIdx.x;
    int C4 = 4 * cin;
    int tot = Bn * Ho * Wo * C4;
    if (idx >= tot) return;
    int c4  = idx % C4;
    int pos = idx / C4;
    int w   = pos % Wo;
    int h   = (pos / Wo) % Ho;
    int b   = pos / (Wo * Ho);
    int q = c4 / cin, c = c4 % cin;
    int dh = q & 1, dw = q >> 1;          // x0:(0,0) x1:(1,0) x2:(0,1) x3:(1,1)
    y4[idx] = x[(((size_t)b * (2 * Ho) + (2 * h + dh)) * (2 * Wo) + (2 * w + dw)) * cin + c];
}

// =========================== row LayerNorm ===========================
__global__ void k_rowln(const float* __restrict__ in, float* __restrict__ out,
                        const float* __restrict__ g, const float* __restrict__ be,
                        int rows, int C)
{
    int row = blockIdx.x * blockDim.x + threadIdx.x;
    if (row >= rows) return;
    const float* p = in + (size_t)row * C;
    float m = 0.f;
    for (int c = 0; c < C; ++c) m += p[c];
    m /= C;
    float v = 0.f;
    for (int c = 0; c < C; ++c) { float d = p[c] - m; v += d * d; }
    v /= C;
    float rinv = rsqrtf(v + 1e-5f);
    float* q = out + (size_t)row * C;
    for (int c = 0; c < C; ++c) q[c] = (p[c] - m) * rinv * g[c] + be[c];
}

// ================= depthwise 3x3 conv + SiLU, emit xf & xt ===========
__global__ void k_dwconv(const float* __restrict__ xz, const float* __restrict__ cw,
                         const float* __restrict__ cb,
                         float* __restrict__ xf, float* __restrict__ xt,
                         int Bn, int H, int Wd, int di)
{
    int idx = blockIdx.x * blockDim.x + threadIdx.x;
    int tot = Bn * H * Wd * di;
    if (idx >= tot) return;
    int d = idx % di;
    int w = (idx / di) % Wd;
    int h = (idx / (di * Wd)) % H;
    int b = idx / (di * Wd * H);
    int L = H * Wd, s2 = 2 * di;
    float acc = cb[d];
    for (int kh = 0; kh < 3; ++kh)
        for (int kw = 0; kw < 3; ++kw) {
            int h2 = h + kh - 1, w2 = w + kw - 1;
            if (h2 >= 0 && h2 < H && w2 >= 0 && w2 < Wd)
                acc += xz[((size_t)b * L + h2 * Wd + w2) * s2 + d] * cw[(kh * 3 + kw) * di + d];
        }
    float s = acc / (1.0f + expf(-acc));      // SiLU
    xf[((size_t)b * L + h * Wd + w) * di + d] = s;
    xt[((size_t)b * L + w * H  + h) * di + d] = s;
}

// ================== dt = softplus(xdbl[..,:R] @ dtw^T + dtb) =========
__global__ void k_dt(const float* __restrict__ xdbl, const float* __restrict__ dtw,
                     const float* __restrict__ dtbW, float* __restrict__ dt,
                     int Bn, int L, int di, int R, int Cc)
{
    int idx = blockIdx.x * blockDim.x + threadIdx.x;
    int tot = 4 * Bn * L * di;
    if (idx >= tot) return;
    int d = idx % di;
    int l = (idx / di) % L;
    int b = (idx / (di * L)) % Bn;
    int k = idx / (di * L * Bn);
    const float* xr = xdbl + ((size_t)(k * Bn + b) * L + l) * Cc;
    float v = dtbW[k * di + d];
    for (int r = 0; r < R; ++r) v += xr[r] * dtw[(k * di + d) * R + r];
    dt[idx] = (v > 20.0f) ? v : log1pf(expf(v));
}

// ============== chunked selective scan: pass 1 (local) ===============
// 16 lanes per (k,b,d,chunk); chunk covers scan-time [c*CS, c*CS+len).
// Produces composite (prod a, local b) per (k,b,d,chunk,n).
__global__ void k_scan_part1(const float* __restrict__ xdbl, const float* __restrict__ dtB,
                             const float* __restrict__ xf, const float* __restrict__ xt,
                             const float* __restrict__ alog,
                             float* __restrict__ chA, float* __restrict__ chB,
                             int Bn, int L, int di, int R, int Cc, int NC)
{
    int t = blockIdx.x * blockDim.x + threadIdx.x;
    int g = t >> 4, n = t & 15;
    if (g >= 4 * Bn * di * NC) return;
    int c  = g % NC;
    int g2 = g / NC;
    int k = g2 / (Bn * di);
    int rem = g2 % (Bn * di);
    int b = rem / di, d = rem % di;

    float An = -expf(alog[((size_t)k * di + d) * 16 + n]);
    const float* xs = (k & 1) ? xt : xf;
    int t0  = c * SCAN_CS;
    int len = (L - t0 < SCAN_CS) ? (L - t0) : SCAN_CS;
    int dir = (k >= 2) ? -1 : 1;
    int l   = (k >= 2) ? (L - 1 - t0) : t0;
    const size_t rowB = (size_t)(k * Bn + b) * L;

    float aP = 1.0f, h = 0.0f;
    for (int s = 0; s < len; ++s, l += dir) {
        float dtv = dtB[(rowB + l) * di + d];
        float xv  = xs[((size_t)b * L + l) * di + d];
        float Bv  = xdbl[(rowB + l) * Cc + R + n];
        float a   = expf(dtv * An);
        h  = fmaf(a, h, dtv * xv * Bv);
        aP *= a;
    }
    size_t ci = ((((size_t)(k * Bn + b) * di + d) * NC) + c) * 16 + n;
    chA[ci] = aP;
    chB[ci] = h;
}

// ============ chunked scan: pass 2 (combine chunk summaries) =========
// One thread per (k,b,d,n): sequential over NC chunk summaries; writes
// each chunk's initial state.
__global__ void k_scan_part2(const float* __restrict__ chA, const float* __restrict__ chB,
                             float* __restrict__ hInit, int Bn, int di, int NC)
{
    int idx = blockIdx.x * blockDim.x + threadIdx.x;
    int tot = 4 * Bn * di * 16;
    if (idx >= tot) return;
    int n = idx & 15;
    int g = idx >> 4;                 // (k*Bn+b)*di + d
    size_t base = (size_t)g * NC * 16 + n;
    float s = 0.0f;
    for (int c = 0; c < NC; ++c) {
        size_t ci = base + (size_t)c * 16;
        hInit[ci] = s;
        s = chA[ci] * s + chB[ci];
    }
}

// ============ chunked scan: pass 3 (rescan + output) =================
// Same mapping as pass 1; h starts from hInit; C-projection via
// __shfl_xor 16-lane reduce; writes ys (k>=2 stored pre-flipped).
__global__ void k_scan_part3(const float* __restrict__ xdbl, const float* __restrict__ dtB,
                             const float* __restrict__ xf, const float* __restrict__ xt,
                             const float* __restrict__ alog, const float* __restrict__ Ds,
                             const float* __restrict__ hInit, float* __restrict__ ys,
                             int Bn, int L, int di, int R, int Cc, int NC)
{
    int t = blockIdx.x * blockDim.x + threadIdx.x;
    int g = t >> 4, n = t & 15;
    if (g >= 4 * Bn * di * NC) return;
    int c  = g % NC;
    int g2 = g / NC;
    int k = g2 / (Bn * di);
    int rem = g2 % (Bn * di);
    int b = rem / di, d = rem % di;

    float An = -expf(alog[((size_t)k * di + d) * 16 + n]);
    float Dv = Ds[k * di + d];
    const float* xs = (k & 1) ? xt : xf;
    int t0  = c * SCAN_CS;
    int len = (L - t0 < SCAN_CS) ? (L - t0) : SCAN_CS;
    int dir = (k >= 2) ? -1 : 1;
    int l   = (k >= 2) ? (L - 1 - t0) : t0;
    const size_t rowB = (size_t)(k * Bn + b) * L;

    size_t ci = ((((size_t)(k * Bn + b) * di + d) * NC) + c) * 16 + n;
    float h = hInit[ci];
    for (int s = 0; s < len; ++s, l += dir) {
        float dtv = dtB[(rowB + l) * di + d];
        float xv  = xs[((size_t)b * L + l) * di + d];
        const float* xr = xdbl + (rowB + l) * Cc + R;
        float Bv = xr[n];
        float Cv = xr[16 + n];
        float a  = expf(dtv * An);
        h = fmaf(a, h, dtv * xv * Bv);
        float part = h * Cv;
        part += __shfl_xor(part, 8);
        part += __shfl_xor(part, 4);
        part += __shfl_xor(part, 2);
        part += __shfl_xor(part, 1);
        if (n == 0) ys[(rowB + l) * di + d] = part + Dv * xv;
    }
}

// ====== y_sum = ys0 + flip(ys2) + wh(ys1) + wh(flip(ys3)) ============
__global__ void k_ysum(const float* __restrict__ ys, float* __restrict__ ysum,
                       int Bn, int H, int Wd, int di)
{
    int idx = blockIdx.x * blockDim.x + threadIdx.x;
    int L = H * Wd;
    int tot = Bn * L * di;
    if (idx >= tot) return;
    int d = idx % di;
    int l = (idx / di) % L;
    int b = idx / (di * L);
    int h = l / Wd, w = l % Wd;
    int lt = w * H + h;
    float v = ys[((size_t)(0 * Bn + b) * L + l)  * di + d]
            + ys[((size_t)(2 * Bn + b) * L + l)  * di + d]
            + ys[((size_t)(1 * Bn + b) * L + lt) * di + d]
            + ys[((size_t)(3 * Bn + b) * L + lt) * di + d];
    ysum[((size_t)b * L + l) * di + d] = v;
}

// ============ gated = LayerNorm(ysum; ong,onb) * SiLU(z) =============
__global__ void k_gate(const float* __restrict__ ysum, const float* __restrict__ xz,
                       const float* __restrict__ g, const float* __restrict__ be,
                       float* __restrict__ gated, int rows, int di)
{
    int row = blockIdx.x * blockDim.x + threadIdx.x;
    if (row >= rows) return;
    const float* p = ysum + (size_t)row * di;
    float m = 0.f;
    for (int c = 0; c < di; ++c) m += p[c];
    m /= di;
    float v = 0.f;
    for (int c = 0; c < di; ++c) { float d0 = p[c] - m; v += d0 * d0; }
    v /= di;
    float rinv = rsqrtf(v + 1e-5f);
    const float* zrow = xz + (size_t)row * (2 * di) + di;
    float* q = gated + (size_t)row * di;
    for (int c = 0; c < di; ++c) {
        float z = zrow[c];
        q[c] = ((p[c] - m) * rinv * g[c] + be[c]) * (z / (1.0f + expf(-z)));
    }
}

// ======================== head: pool + linear ========================
__global__ void k_pool(const float* __restrict__ hln, float* __restrict__ pooled,
                       int Bn, int C)
{
    int idx = blockIdx.x * blockDim.x + threadIdx.x;
    if (idx >= Bn * C) return;
    int c = idx % C, b = idx / C;
    float s = 0.f;
    for (int p = 0; p < 4; ++p) s += hln[((size_t)b * 4 + p) * C + c];
    pooled[idx] = s * 0.25f;
}

__global__ void k_head(const float* __restrict__ pooled, const float* __restrict__ w,
                       const float* __restrict__ bh, float* __restrict__ out,
                       int Bn, int C, int NC)
{
    int idx = blockIdx.x * blockDim.x + threadIdx.x;
    if (idx >= Bn * NC) return;
    int n = idx % NC, b = idx / NC;
    float s = bh[n];
    for (int c = 0; c < C; ++c) s += pooled[(size_t)b * C + c] * w[(size_t)n * C + c];
    out[(size_t)b * NC + n] = s;
}

// =====================================================================
static inline dim3 g1(int tot, int bs) { return dim3((tot + bs - 1) / bs); }

extern "C" void kernel_launch(void* const* d_in, const int* in_sizes, int n_in,
                              void* d_out, int out_size, void* d_ws, size_t ws_size,
                              hipStream_t stream)
{
    (void)in_sizes; (void)n_in; (void)out_size; (void)ws_size;
    static const int DIMS[7] = {3, 6, 12, 24, 48, 96, 192};
    const int B = 8;

    // ---- workspace carve (floats; each chunk 16B-aligned) ----
    float* ws = (float*)d_ws;
    size_t o = 0;
    auto alloc = [&](size_t n) { float* p = ws + o; o += n; return p; };
    float* bufA   = alloc(393216);   // stem out / stage ping
    float* bufB   = alloc(393216);   // stage pong
    float* y4     = alloc(393216);   // merge concat (+LN in place)
    float* merged = alloc(196608);   // stage residual base
    float* yln    = alloc(196608);
    float* xz     = alloc(786432);   // (B*L, 2*di)
    float* xf     = alloc(393216);
    float* xt     = alloc(393216);
    float* xdbl   = alloc(4325376);  // (4,B,L,Cc)
    float* dtBuf  = alloc(1572864);  // (4,B,L,di)
    float* ysBuf  = alloc(1572864);  // (4,B,L,di)
    float* ysum   = alloc(393216);
    float* gated  = alloc(393216);
    float* chA    = alloc(393216);   // (4,B,di,NC,16) chunk prod-a
    float* chB    = alloc(393216);   // (4,B,di,NC,16) chunk local-b
    float* hInit  = alloc(393216);   // (4,B,di,NC,16) chunk init state
    float* hln    = alloc(8 * 4 * 192);
    float* pooled = alloc(8 * 192);

    const float* xin = (const float*)d_in[0];
    const float* stw = (const float*)d_in[1];
    const float* stb = (const float*)d_in[2];

    // ---- stem conv + ReLU -> NHWC ----
    {
        int tot = B * 128 * 128 * 3;
        k_stem<<<g1(tot, 256), 256, 0, stream>>>(xin, stw, stb, bufA, B, 128, 128);
    }

    float* cur = bufA;
    float* nxt = bufB;
    int Hin = 128;

    for (int i = 0; i < 6; ++i) {
        const int cin = DIMS[i], cout = DIMS[i + 1];
        const int di = 2 * cout;
        const int R = (cout + 15) / 16;
        const int Cc = R + 32;
        const int Ho = Hin / 2, Wo = Ho, L = Ho * Wo;
        const int M = B * L;
        const int NC = (L + SCAN_CS - 1) / SCAN_CS;
        const int base = 3 + i * 16;
        const float* mg   = (const float*)d_in[base + 0];
        const float* mb   = (const float*)d_in[base + 1];
        const float* mw   = (const float*)d_in[base + 2];
        const float* lg   = (const float*)d_in[base + 3];
        const float* lb   = (const float*)d_in[base + 4];
        const float* inw  = (const float*)d_in[base + 5];
        const float* cw   = (const float*)d_in[base + 6];
        const float* cb   = (const float*)d_in[base + 7];
        const float* xp   = (const float*)d_in[base + 8];
        const float* dtw  = (const float*)d_in[base + 9];
        const float* dtbW = (const float*)d_in[base + 10];
        const float* alog = (const float*)d_in[base + 11];
        const float* Ds   = (const float*)d_in[base + 12];
        const float* ong  = (const float*)d_in[base + 13];
        const float* onb  = (const float*)d_in[base + 14];
        const float* outw = (const float*)d_in[base + 15];

        // patch merge: concat -> LN -> GEMM(mw)
        k_pmerge<<<g1(M * 4 * cin, 256), 256, 0, stream>>>(cur, y4, B, Ho, Wo, cin);
        k_rowln<<<g1(M, 256), 256, 0, stream>>>(y4, y4, mg, mb, M, 4 * cin);
        {
            dim3 gg((M + 127) / 128, (cout + 15) / 16);
            k_gemm_wmma<<<gg, 256, 0, stream>>>(y4, mw, nullptr, nullptr, merged, M, cout, 4 * cin);
        }

        // VSS block
        k_rowln<<<g1(M, 256), 256, 0, stream>>>(merged, yln, lg, lb, M, cout);
        {
            dim3 gg((M + 127) / 128, (2 * di + 15) / 16);
            k_gemm_wmma<<<gg, 256, 0, stream>>>(yln, inw, nullptr, nullptr, xz, M, 2 * di, cout);
        }
        k_dwconv<<<g1(M * di, 256), 256, 0, stream>>>(xz, cw, cb, xf, xt, B, Ho, Wo, di);

        for (int k = 0; k < 4; ++k) {
            const float* xs = (k & 1) ? xt : xf;
            dim3 gg((M + 127) / 128, (Cc + 15) / 16);
            k_gemm_wmma<<<gg, 256, 0, stream>>>(xs, xp + (size_t)k * Cc * di, nullptr, nullptr,
                                                xdbl + (size_t)k * M * Cc, M, Cc, di);
        }

        k_dt<<<g1(4 * M * di, 256), 256, 0, stream>>>(xdbl, dtw, dtbW, dtBuf, B, L, di, R, Cc);

        // three-pass chunked selective scan
        k_scan_part1<<<g1(4 * B * di * NC * 16, 128), 128, 0, stream>>>(
            xdbl, dtBuf, xf, xt, alog, chA, chB, B, L, di, R, Cc, NC);
        k_scan_part2<<<g1(4 * B * di * 16, 256), 256, 0, stream>>>(
            chA, chB, hInit, B, di, NC);
        k_scan_part3<<<g1(4 * B * di * NC * 16, 128), 128, 0, stream>>>(
            xdbl, dtBuf, xf, xt, alog, Ds, hInit, ysBuf, B, L, di, R, Cc, NC);

        k_ysum<<<g1(M * di, 256), 256, 0, stream>>>(ysBuf, ysum, B, Ho, Wo, di);
        k_gate<<<g1(M, 256), 256, 0, stream>>>(ysum, xz, ong, onb, gated, M, di);
        {
            dim3 gg((M + 127) / 128, (cout + 15) / 16);
            k_gemm_wmma<<<gg, 256, 0, stream>>>(gated, outw, nullptr, merged, nxt, M, cout, di);
        }

        float* tmp = cur; cur = nxt; nxt = tmp;
        Hin = Ho;
    }

    // ---- head: LN -> mean pool -> linear ----
    const float* hg  = (const float*)d_in[99];
    const float* hb  = (const float*)d_in[100];
    const float* hw  = (const float*)d_in[101];
    const float* hbh = (const float*)d_in[102];
    k_rowln<<<g1(B * 4, 64), 64, 0, stream>>>(cur, hln, hg, hb, B * 4, 192);
    k_pool<<<g1(B * 192, 256), 256, 0, stream>>>(hln, pooled, B, 192);
    k_head<<<g1(B * 43, 256), 256, 0, stream>>>(pooled, hw, hbh, (float*)d_out, B, 192, 43);
}
